// MultiheadAttentionRelativeEncoding_46557445488844
// MI455X (gfx1250) — matrix operand order repaired and checked by reading, not compiled, not run on hardware
//
#include <hip/hip_runtime.h>
#include <stdint.h>

// Problem constants (reference: H=8, D=512, S=2048, B=2, DH=64, REL_CLIP=2048)
enum { kH = 8, kB = 2, kS = 2048, kD = 512, kDH = 64, kQB = 64 };

typedef __bf16 bf16_t;
typedef bf16_t v16bf __attribute__((ext_vector_type(16)));
typedef float  v8f   __attribute__((ext_vector_type(8)));

union FragBF { v16bf v; uint4 q[2]; unsigned short s[16]; };

__device__ __forceinline__ unsigned short f2bf(float f) {
  union { float f; uint32_t u; } v; v.f = f;
  uint32_t r = v.u + 0x7FFFu + ((v.u >> 16) & 1u);   // round-to-nearest-even
  return (unsigned short)(r >> 16);
}
__device__ __forceinline__ float bf2f(unsigned short b) {
  union { uint32_t u; float f; } v; v.u = ((uint32_t)b) << 16; return v.f;
}
__device__ __forceinline__ v8f v8zero() {
  v8f z;
  #pragma unroll
  for (int i = 0; i < 8; ++i) z[i] = 0.0f;
  return z;
}

// A fragment (16x32 bf16, row-major source, lane holds row m=lane&15):
// elems 0..7  -> k = kbase + 8*(lane>=16) + e
// elems 8..15 -> k = kbase + 16 + 8*(lane>=16) + (e-8)
__device__ __forceinline__ v16bf loadA(const unsigned short* p, int m, int lda,
                                       int kbase, int lane) {
  const int off = m * lda + kbase + ((lane >> 4) << 3);
  FragBF f;
  f.q[0] = *(const uint4*)(p + off);
  f.q[1] = *(const uint4*)(p + off + 16);
  return f.v;
}
// B fragment (32x16 bf16) from n-major storage Bt[n][k] (lane holds col n=lane&15):
// elems 0..7  -> k = kbase + 8*(lane>=16) + e
// elems 8..15 -> k = kbase + 16 + 8*(lane>=16) + (e-8)
__device__ __forceinline__ v16bf loadB(const unsigned short* p, int n, int ldb,
                                       int kbase, int lane) {
  const int off = n * ldb + kbase + ((lane >> 4) << 3);
  FragBF f;
  f.q[0] = *(const uint4*)(p + off);
  f.q[1] = *(const uint4*)(p + off + 16);
  return f.v;
}
// A fragment built from f32 source (on-the-fly bf16 convert)
__device__ __forceinline__ v16bf loadA_f32(const float* p, int m, int lda,
                                           int kbase, int lane) {
  const int koff = (lane >> 4) << 3;
  const float* row = p + m * lda + kbase;
  FragBF f;
  #pragma unroll
  for (int e = 0; e < 8; ++e) f.s[e]     = f2bf(row[koff + e]);
  #pragma unroll
  for (int e = 0; e < 8; ++e) f.s[8 + e] = f2bf(row[16 + koff + e]);
  return f.v;
}

__device__ __forceinline__ v8f wmma_bf16(v16bf a, v16bf b, v8f c) {
  return __builtin_amdgcn_wmma_f32_16x16x32_bf16(false, a, false, b,
                                                 (short)0, c, false, false);
}

// ---------------------------------------------------------------------------
// Prep: bf16 transposes of Wq/Wk/Wv -> Wt[3][H][DH][D], Wo -> Wob[D][D],
//       att_rel_emb -> Erb[H][S][DH]
// ---------------------------------------------------------------------------
__global__ void prep_kernel(const float* __restrict__ Wq, const float* __restrict__ Wk,
                            const float* __restrict__ Wv, const float* __restrict__ Wo,
                            const float* __restrict__ Er,
                            unsigned short* __restrict__ Wt,
                            unsigned short* __restrict__ Wob,
                            unsigned short* __restrict__ Erb) {
  const int nW  = 3 * kH * kDH * kD;
  const int nWo = kD * kD;
  const int nEr = kH * kS * kDH;
  int idx = blockIdx.x * blockDim.x + threadIdx.x;
  if (idx < nW) {
    int m = idx % kD; int r = idx / kD;
    int n = r % kDH;  r /= kDH;
    int h = r % kH;   int w = r / kH;
    const float* W = (w == 0) ? Wq : ((w == 1) ? Wk : Wv);
    Wt[idx] = f2bf(W[((size_t)h * kD + m) * kDH + n]);
  } else if (idx < nW + nWo) {
    int i = idx - nW;
    int m = i % kD, n = i / kD;
    Wob[(size_t)n * kD + m] = f2bf(Wo[(size_t)m * kD + n]);
  } else if (idx < nW + nWo + nEr) {
    int i = idx - nW - nWo;
    int d = i % kDH; int r = i / kDH;
    int t = r % kS;  int h = r / kS;
    // REL_CLIP == S so the last-S slice is the whole thing
    Erb[i] = f2bf(Er[((size_t)h * kDH + d) * kS + t]);
  }
}

// ---------------------------------------------------------------------------
// QKV projection: Qb/Kb [h][b*S][DH] bf16 (Q pre-scaled), Vtb [h][b][DH][S] bf16
// grid (B*S/64, H, 3), 128 threads (4 waves), wave -> 16 rows x 64 cols
// ---------------------------------------------------------------------------
__global__ void __launch_bounds__(128)
proj_kernel(const float* __restrict__ X, const unsigned short* __restrict__ Wt,
            unsigned short* __restrict__ Qb, unsigned short* __restrict__ Kb,
            unsigned short* __restrict__ Vtb) {
  const int lane = threadIdx.x & 31, wave = threadIdx.x >> 5;
  const int half = lane >> 4, l16 = lane & 15;
  const int which = blockIdx.z;
  const int h = blockIdx.y;
  const int row0 = blockIdx.x * 64 + wave * 16;     // flat row over b*S+s
  const unsigned short* Wth = Wt + ((size_t)which * kH + h) * kDH * kD;

  v8f acc[4] = { v8zero(), v8zero(), v8zero(), v8zero() };
  for (int kk = 0; kk < kD; kk += 32) {
    v16bf a = loadA_f32(X, row0 + l16, kD, kk, lane);
    #pragma unroll
    for (int nt = 0; nt < 4; ++nt)
      acc[nt] = wmma_bf16(a, loadB(Wth, nt * 16 + l16, kD, kk, lane), acc[nt]);
  }
  const float scale = (which == 0) ? 0.125f : 1.0f;  // 1/sqrt(DH)
  #pragma unroll
  for (int nt = 0; nt < 4; ++nt) {
    #pragma unroll
    for (int r = 0; r < 8; ++r) {
      const int row = row0 + r + 8 * half;          // flat b*S+s
      const int col = nt * 16 + l16;
      const unsigned short v = f2bf(acc[nt][r] * scale);
      if (which == 0)
        Qb[((size_t)h * (kB * kS) + row) * kDH + col] = v;
      else if (which == 1)
        Kb[((size_t)h * (kB * kS) + row) * kDH + col] = v;
      else {
        const int b = row / kS, s = row % kS;
        Vtb[(((size_t)h * kB + b) * kDH + col) * kS + s] = v;
      }
    }
  }
}

// ---------------------------------------------------------------------------
// Fused attention with relative skew, flash-style online softmax.
// grid (S/64, H*B), 128 threads (4 waves), wave -> 16 query rows.
// LDS: rel panel 64 x 2048 bf16 (256KB, fits CDNA5 320KB WGP LDS) + P staging.
// ---------------------------------------------------------------------------
#define SMEM_USHORTS (kQB * kS + 4 * 16 * 32)
#define SMEM_BYTES   (SMEM_USHORTS * 2)

__device__ __forceinline__ float skew_val(const unsigned short* rel, int ql,
                                          int qg, int c) {
  // Srel[qg][c] per music-transformer skew of rel (incl. wrapped upper tri)
  if (c <= qg) return bf2f(rel[ql * kS + (kS - 1 - qg + c)]);
  if (c == qg + 1) return 0.0f;
  const int ql2 = ql + 1;
  if (ql2 >= kQB) return 0.0f;     // block-boundary row
  return bf2f(rel[ql2 * kS + (c - qg - 2)]);
}

__global__ void __launch_bounds__(128)
attn_kernel(const unsigned short* __restrict__ Qb, const unsigned short* __restrict__ Kb,
            const unsigned short* __restrict__ Vtb, const unsigned short* __restrict__ Erb,
            const float* __restrict__ mask, unsigned short* __restrict__ AttO) {
  extern __shared__ unsigned short smem[];          // [64][S] rel bf16 + P tiles
  const int lane = threadIdx.x & 31, wave = threadIdx.x >> 5;
  const int half = lane >> 4, l16 = lane & 15;
  const int hb = blockIdx.y;                        // h*B + b
  const int h = hb / kB, b = hb % kB;
  const int q0 = blockIdx.x * kQB;
  const unsigned short* Qhb = Qb + (size_t)hb * kS * kDH;
  const unsigned short* Khb = Kb + (size_t)hb * kS * kDH;
  const unsigned short* Vhb = Vtb + (size_t)hb * kDH * kS;
  const unsigned short* Erh = Erb + (size_t)h * kS * kDH;
  const int rw = wave * 16;                         // wave's local row base

  // Q fragments for this wave's 16 rows (reused in both phases)
  const v16bf qa0 = loadA(Qhb, q0 + rw + l16, kDH, 0, lane);
  const v16bf qa1 = loadA(Qhb, q0 + rw + l16, kDH, 32, lane);

  // ---- Phase 1: rel[ql][t] = Q[q0+ql,:] . Er[t,:]  -> LDS panel
  for (int t0 = 0; t0 < kS; t0 += 16) {
    v8f c = v8zero();
    c = wmma_bf16(qa0, loadB(Erh, t0 + l16, kDH, 0, lane), c);
    c = wmma_bf16(qa1, loadB(Erh, t0 + l16, kDH, 32, lane), c);
    #pragma unroll
    for (int r = 0; r < 8; ++r)
      smem[(rw + r + 8 * half) * kS + t0 + l16] = f2bf(c[r]);
  }
  __syncthreads();

  // ---- Phase 2: streaming softmax(QK^T + Srel + mask) V
  unsigned short* Pl = smem + kQB * kS + wave * (16 * 32);
  float mrow[8], lrow[8];
  #pragma unroll
  for (int r = 0; r < 8; ++r) { mrow[r] = -1e30f; lrow[r] = 0.0f; }
  v8f O[4] = { v8zero(), v8zero(), v8zero(), v8zero() };

  for (int t0 = 0; t0 < kS; t0 += 32) {
    if (t0 + 32 < kS) {                             // global_prefetch_b8
      __builtin_prefetch(Khb + (size_t)(t0 + 32) * kDH, 0, 1);
      __builtin_prefetch(Vhb + (size_t)l16 * kS + t0 + 32, 0, 1);
    }
    v8f s0 = v8zero(), s1 = v8zero();
    s0 = wmma_bf16(qa0, loadB(Khb, t0 + l16, kDH, 0, lane), s0);
    s0 = wmma_bf16(qa1, loadB(Khb, t0 + l16, kDH, 32, lane), s0);
    s1 = wmma_bf16(qa0, loadB(Khb, t0 + 16 + l16, kDH, 0, lane), s1);
    s1 = wmma_bf16(qa1, loadB(Khb, t0 + 16 + l16, kDH, 32, lane), s1);

    const int c0 = t0 + l16, c1 = t0 + 16 + l16;
    #pragma unroll
    for (int r = 0; r < 8; ++r) {
      const int ql = rw + r + 8 * half;
      const int qg = q0 + ql;
      s0[r] += skew_val(smem, ql, qg, c0) + mask[(size_t)qg * kS + c0];
      s1[r] += skew_val(smem, ql, qg, c1) + mask[(size_t)qg * kS + c1];
    }
    // online softmax (row reductions across 16-lane groups; wave32 shuffles)
    #pragma unroll
    for (int r = 0; r < 8; ++r) {
      float x = fmaxf(s0[r], s1[r]);
      #pragma unroll
      for (int i = 1; i < 16; i <<= 1) x = fmaxf(x, __shfl_xor(x, i, 32));
      const float mn = fmaxf(mrow[r], x);
      const float ef = __expf(mrow[r] - mn);
      const float p0 = __expf(s0[r] - mn);
      const float p1 = __expf(s1[r] - mn);
      float sum = p0 + p1;
      #pragma unroll
      for (int i = 1; i < 16; i <<= 1) sum += __shfl_xor(sum, i, 32);
      lrow[r] = lrow[r] * ef + sum;
      mrow[r] = mn;
      O[0][r] *= ef; O[1][r] *= ef; O[2][r] *= ef; O[3][r] *= ef;
      Pl[(r + 8 * half) * 32 + l16]      = f2bf(p0);   // C-layout -> LDS
      Pl[(r + 8 * half) * 32 + 16 + l16] = f2bf(p1);
    }
    // P (A-layout from per-wave LDS tile; intra-wave DS ops are in-order) x V
    const v16bf pa = loadA(Pl, l16, 32, 0, lane);
    #pragma unroll
    for (int dt = 0; dt < 4; ++dt)
      O[dt] = wmma_bf16(pa, loadB(Vhb, dt * 16 + l16, kS, t0, lane), O[dt]);
  }

  // normalize + store bf16 into AttO [b][s][h*DH + d]
  #pragma unroll
  for (int r = 0; r < 8; ++r) {
    const float inv = 1.0f / lrow[r];
    const int s = q0 + rw + r + 8 * half;
    #pragma unroll
    for (int dt = 0; dt < 4; ++dt) {
      const int col = h * kDH + dt * 16 + l16;
      AttO[((size_t)b * kS + s) * kD + col] = f2bf(O[dt][r] * inv);
    }
  }
}

// ---------------------------------------------------------------------------
// Output GEMM: out[B*S][D] = AttO(bf16) x Wo ; Wob pre-transposed [n][k]
// ---------------------------------------------------------------------------
__global__ void __launch_bounds__(128)
out_gemm(const unsigned short* __restrict__ AttO, const unsigned short* __restrict__ Wob,
         float* __restrict__ out) {
  const int lane = threadIdx.x & 31, wave = threadIdx.x >> 5;
  const int half = lane >> 4, l16 = lane & 15;
  const int row0 = blockIdx.x * 64 + wave * 16;
  const int cbase = blockIdx.y * 64;

  v8f acc[4] = { v8zero(), v8zero(), v8zero(), v8zero() };
  for (int kk = 0; kk < kD; kk += 32) {
    v16bf a = loadA(AttO, row0 + l16, kD, kk, lane);
    #pragma unroll
    for (int nt = 0; nt < 4; ++nt)
      acc[nt] = wmma_bf16(a, loadB(Wob, cbase + nt * 16 + l16, kD, kk, lane),
                          acc[nt]);
  }
  #pragma unroll
  for (int nt = 0; nt < 4; ++nt)
    #pragma unroll
    for (int r = 0; r < 8; ++r)
      out[(size_t)(row0 + r + 8 * half) * kD + cbase + nt * 16 + l16] = acc[nt][r];
}

// ---------------------------------------------------------------------------
extern "C" void kernel_launch(void* const* d_in, const int* in_sizes, int n_in,
                              void* d_out, int out_size, void* d_ws, size_t ws_size,
                              hipStream_t stream) {
  (void)in_sizes; (void)n_in; (void)out_size; (void)ws_size;
  const float* X    = (const float*)d_in[0];
  const float* mask = (const float*)d_in[1];
  const float* Wq   = (const float*)d_in[2];
  const float* Wk   = (const float*)d_in[3];
  const float* Wv   = (const float*)d_in[4];
  const float* Wo   = (const float*)d_in[5];
  const float* Er   = (const float*)d_in[6];
  float* out = (float*)d_out;

  char* ws = (char*)d_ws;                               // ~20 MB total
  unsigned short* Wt   = (unsigned short*)(ws + 0);         // 3*8*64*512*2 = 1.5MB
  unsigned short* Wob  = (unsigned short*)(ws + 1572864);   // 512*512*2    = 0.5MB
  unsigned short* Erb  = (unsigned short*)(ws + 2097152);   // 8*2048*64*2  = 2MB
  unsigned short* Qb   = (unsigned short*)(ws + 4194304);   // 4MB
  unsigned short* Kb   = (unsigned short*)(ws + 8388608);   // 4MB
  unsigned short* Vtb  = (unsigned short*)(ws + 12582912);  // 4MB
  unsigned short* AttO = (unsigned short*)(ws + 16777216);  // 4MB

  const int prepN = 3 * kH * kDH * kD + kD * kD + kH * kS * kDH;
  prep_kernel<<<(prepN + 255) / 256, 256, 0, stream>>>(Wq, Wk, Wv, Wo, Er,
                                                       Wt, Wob, Erb);
  proj_kernel<<<dim3(kB * kS / 64, kH, 3), 128, 0, stream>>>(X, Wt, Qb, Kb, Vtb);

  (void)hipFuncSetAttribute(reinterpret_cast<const void*>(attn_kernel),
                            hipFuncAttributeMaxDynamicSharedMemorySize, SMEM_BYTES);
  attn_kernel<<<dim3(kS / kQB, kH * kB), 128, SMEM_BYTES, stream>>>(
      Qb, Kb, Vtb, Erb, mask, AttO);

  out_gemm<<<dim3(kB * kS / 64, kD / 64), 128, 0, stream>>>(AttO, Wob, out);
}
